// NearestConvolution_40862318854437
// MI455X (gfx1250) — matrix operand order repaired and checked by reading, not compiled, not run on hardware
//
#include <hip/hip_runtime.h>
#include <hip/hip_bf16.h>

// NearestConvolution on MI455X (gfx1250, wave32, WMMA).
// Pipeline: normalize -> fused WMMA Gram + streaming top-3 -> conv/softmax/pool -> WMMA FC+ReLU.
// Round-2 changes: 4-wide column blocking per K-step (ILP-4 WMMA chains), rolled K loop to
// kill the LICM-induced scratch spills, 2x wave split over column space for occupancy.

typedef __attribute__((ext_vector_type(16))) _Float16 v16h;
typedef __attribute__((ext_vector_type(8)))  _Float16 v8h;
typedef __attribute__((ext_vector_type(8)))  float    v8f;

#define BATCH 8
#define NPTS  2048
#define DIM   512
#define KNN   3
#define DOUT  512

// ---------------- kernel 1: per-row L2 normalize, emit f16 ----------------
__global__ void k_normalize(const float* __restrict__ feats,
                            _Float16* __restrict__ xnh) {
  int wave = (int)((blockIdx.x * blockDim.x + threadIdx.x) >> 5);
  int lane = threadIdx.x & 31;
  if (wave >= BATCH * NPTS) return;
  const float* row = feats + (size_t)wave * DIM + lane * 16;
  float v[16];
  float s = 0.f;
  #pragma unroll
  for (int t = 0; t < 16; ++t) { v[t] = row[t]; s += v[t] * v[t]; }
  #pragma unroll
  for (int off = 16; off >= 1; off >>= 1) s += __shfl_xor(s, off, 32);
  float inv = 1.f / fmaxf(sqrtf(s), 1e-12f);
  _Float16* orow = xnh + (size_t)wave * DIM + lane * 16;
  #pragma unroll
  for (int t = 0; t < 16; ++t) orow[t] = (_Float16)(v[t] * inv);
}

// ---------------- kernel 2: fc weight f32 -> f16 ----------------
__global__ void k_fcw_convert(const float* __restrict__ w,
                              _Float16* __restrict__ wh) {
  int i = blockIdx.x * blockDim.x + threadIdx.x;
  int stride = gridDim.x * blockDim.x;
  for (; i < DOUT * DIM; i += stride) wh[i] = (_Float16)w[i];
}

// ---------------- kernel 3: fused Gram (Xn Xn^T) + streaming top-3 ----------------
// Block = 4 waves = 2 row-strips x 2 column-halves. Each wave processes 64-column
// groups (4 WMMA tiles in flight per K-step), keeps a running per-row top-3 split
// across lane halves, then merges: lane-half merge (shuffle) -> wave merge (LDS).
__global__ void k_sim_topk(const _Float16* __restrict__ xnh,
                           int* __restrict__ knn_idx) {
  __shared__ float tile[4][16][68];   // per-wave 16x64 staging, padded
  __shared__ float mv[2][16][3];      // cross-wave merge values (per strip)
  __shared__ int   mi[2][16][3];      // cross-wave merge indices
  int wv   = threadIdx.x >> 5;
  int lane = threadIdx.x & 31;
  int r    = lane & 15;
  int hi   = lane >> 4;
  int sloc = wv >> 1;                       // strip within block (0..1)
  int colh = wv & 1;                        // column-half owned by this wave
  int strip = blockIdx.x * 2 + sloc;        // global strip 0..1023
  int b    = strip >> 7;                    // / (NPTS/16)
  int m0   = (strip & 127) * 16;
  const _Float16* xb = xnh + (size_t)b * NPTS * DIM;
  // A fragment (16x32 f16): lane holds row m0+r; K {0..15}/{16..31} by lane half.
  const _Float16* aptr = xb + (size_t)(m0 + r) * DIM + hi * 16;

  float tv0 = -1e30f, tv1 = -1e30f, tv2 = -1e30f;
  int   ti0 = 0, ti1 = 0, ti2 = 0;
  auto ins = [&](float v, int id) {
    if (v > tv0)      { tv2=tv1; ti2=ti1; tv1=tv0; ti1=ti0; tv0=v; ti0=id; }
    else if (v > tv1) { tv2=tv1; ti2=ti1; tv1=v; ti1=id; }
    else if (v > tv2) { tv2=v; ti2=id; }
  };

  for (int ctg = colh; ctg < NPTS / 64; ctg += 2) {   // 16 groups per wave
    int c0 = ctg * 64;
    // B fragments (32x16 f16): lane = column, K {0..15}/{16..31} by lane half.
    const _Float16* b0 = xb + (size_t)(c0 + r) * DIM + hi * 16;
    const _Float16* b1 = b0 + (size_t)16 * DIM;
    const _Float16* b2 = b0 + (size_t)32 * DIM;
    const _Float16* b3 = b0 + (size_t)48 * DIM;
    v8f acc0 = {}, acc1 = {}, acc2 = {}, acc3 = {};
    #pragma unroll 2
    for (int k0 = 0; k0 < DIM; k0 += 32) {
      v8h alo = *(const v8h*)(aptr + k0);
      v8h ahi = *(const v8h*)(aptr + k0 + 8);
      v16h a;
      #pragma unroll
      for (int i = 0; i < 8; ++i) { a[i] = alo[i]; a[i + 8] = ahi[i]; }
      acc0 = __builtin_amdgcn_wmma_f32_16x16x32_f16(false, a, false,
                 *(const v16h*)(b0 + k0), (short)0, acc0, false, false);
      acc1 = __builtin_amdgcn_wmma_f32_16x16x32_f16(false, a, false,
                 *(const v16h*)(b1 + k0), (short)0, acc1, false, false);
      acc2 = __builtin_amdgcn_wmma_f32_16x16x32_f16(false, a, false,
                 *(const v16h*)(b2 + k0), (short)0, acc2, false, false);
      acc3 = __builtin_amdgcn_wmma_f32_16x16x32_f16(false, a, false,
                 *(const v16h*)(b3 + k0), (short)0, acc3, false, false);
    }
    // C/D layout: VGPR vr -> row vr (lanes 0-15) / vr+8 (lanes 16-31); col = lane&15.
    #pragma unroll
    for (int vr = 0; vr < 8; ++vr) {
      int rr = vr + 8 * hi;
      tile[wv][rr][     r] = acc0[vr];
      tile[wv][rr][16 + r] = acc1[vr];
      tile[wv][rr][32 + r] = acc2[vr];
      tile[wv][rr][48 + r] = acc3[vr];
    }
    __syncthreads();   // uniform trip count across all 4 waves
    {
      int jb = hi * 32;                 // lane halves scan disjoint column ranges
      #pragma unroll 4
      for (int j = 0; j < 32; ++j) {
        float val = tile[wv][r][jb + j];
        ins(val, c0 + jb + j);
      }
    }
    __syncthreads();
  }

  // merge hi lane-half top-3 into lo lane-half
  float h0 = __shfl(tv0, lane + 16, 32);
  float h1 = __shfl(tv1, lane + 16, 32);
  float h2 = __shfl(tv2, lane + 16, 32);
  int   j0 = __shfl(ti0, lane + 16, 32);
  int   j1 = __shfl(ti1, lane + 16, 32);
  int   j2 = __shfl(ti2, lane + 16, 32);
  if (lane < 16) { ins(h0, j0); ins(h1, j1); ins(h2, j2); }
  __syncthreads();
  if (colh == 1 && lane < 16) {
    mv[sloc][lane][0] = tv0; mv[sloc][lane][1] = tv1; mv[sloc][lane][2] = tv2;
    mi[sloc][lane][0] = ti0; mi[sloc][lane][1] = ti1; mi[sloc][lane][2] = ti2;
  }
  __syncthreads();
  if (colh == 0 && lane < 16) {
    ins(mv[sloc][lane][0], mi[sloc][lane][0]);
    ins(mv[sloc][lane][1], mi[sloc][lane][1]);
    ins(mv[sloc][lane][2], mi[sloc][lane][2]);
    int* o = knn_idx + ((size_t)b * NPTS + m0 + lane) * KNN;
    o[0] = ti0; o[1] = ti1; o[2] = ti2;
  }
}

// ---------------- kernel 4: gather + ConvMapping + softmax + pooled (f16) --------
__global__ void k_conv_pool(const float* __restrict__ feats,
                            const float* __restrict__ cw,
                            const float* __restrict__ cb,
                            const int*   __restrict__ knn_idx,
                            _Float16*    __restrict__ pooled) {
  int wave = (int)((blockIdx.x * blockDim.x + threadIdx.x) >> 5);
  int lane = threadIdx.x & 31;
  if (wave >= BATCH * NPTS) return;
  int b = wave / NPTS;
  const int* ip = knn_idx + (size_t)wave * KNN;
  const float* f0 = feats + ((size_t)b * NPTS + ip[0]) * DIM;
  const float* f1 = feats + ((size_t)b * NPTS + ip[1]) * DIM;
  const float* f2 = feats + ((size_t)b * NPTS + ip[2]) * DIM;
  int e0 = lane * 16;

  float acc[9];
  #pragma unroll
  for (int i = 0; i < 9; ++i) acc[i] = 0.f;
  #pragma unroll 4
  for (int t = 0; t < 16; ++t) {
    int e = e0 + t;
    float nb[3] = { f0[e], f1[e], f2[e] };
    #pragma unroll
    for (int g = 0; g < 3; ++g)
      #pragma unroll
      for (int j = 0; j < 3; ++j)
        acc[g * 3 + j] += nb[g] * cw[(g * 3 + j) * DIM + e];
  }
  #pragma unroll
  for (int i = 0; i < 9; ++i)
    #pragma unroll
    for (int off = 16; off >= 1; off >>= 1)
      acc[i] += __shfl_xor(acc[i], off, 32);

  float wj0 = 0.f, wj1 = 0.f, wj2 = 0.f;
  #pragma unroll
  for (int g = 0; g < 3; ++g) {
    float a0 = acc[g * 3 + 0] + cb[g * 3 + 0];
    float a1 = acc[g * 3 + 1] + cb[g * 3 + 1];
    float a2 = acc[g * 3 + 2] + cb[g * 3 + 2];
    float mx = fmaxf(a0, fmaxf(a1, a2));
    float x0 = __expf(a0 - mx), x1 = __expf(a1 - mx), x2 = __expf(a2 - mx);
    float inv = 1.f / (x0 + x1 + x2);
    wj0 += x0 * inv; wj1 += x1 * inv; wj2 += x2 * inv;
  }
  const float third = 1.f / 3.f;
  wj0 *= third; wj1 *= third; wj2 *= third;

  _Float16* op = pooled + (size_t)wave * DIM + e0;
  #pragma unroll 4
  for (int t = 0; t < 16; ++t) {
    int e = e0 + t;
    op[t] = (_Float16)(wj0 * f0[e] + wj1 * f1[e] + wj2 * f2[e]);
  }
}

// ---------------- kernel 5: out = relu(pooled @ fc_w^T) via f16 WMMA ----------------
// Block = 4 waves = 2 row-tiles x 2 output-halves; 4 output tiles in flight per K-step.
__global__ void k_fc_relu(const _Float16* __restrict__ pooled,
                          const _Float16* __restrict__ fcw,
                          float* __restrict__ out) {
  int wv    = threadIdx.x >> 5;
  int lane  = threadIdx.x & 31;
  int r     = lane & 15;
  int hi    = lane >> 4;
  int mtile = blockIdx.x * 2 + (wv >> 1);   // 0..1023
  int oh    = wv & 1;
  int m0    = mtile * 16;
  const _Float16* aptr = pooled + (size_t)(m0 + r) * DIM + hi * 16;

  for (int otg = oh; otg < DOUT / 64; otg += 2) {   // 4 groups per wave
    int o0 = otg * 64;
    const _Float16* b0 = fcw + (size_t)(o0 + r) * DIM + hi * 16;
    const _Float16* b1 = b0 + (size_t)16 * DIM;
    const _Float16* b2 = b0 + (size_t)32 * DIM;
    const _Float16* b3 = b0 + (size_t)48 * DIM;
    v8f acc0 = {}, acc1 = {}, acc2 = {}, acc3 = {};
    #pragma unroll 2
    for (int k0 = 0; k0 < DIM; k0 += 32) {
      v8h alo = *(const v8h*)(aptr + k0);
      v8h ahi = *(const v8h*)(aptr + k0 + 8);
      v16h a;
      #pragma unroll
      for (int i = 0; i < 8; ++i) { a[i] = alo[i]; a[i + 8] = ahi[i]; }
      acc0 = __builtin_amdgcn_wmma_f32_16x16x32_f16(false, a, false,
                 *(const v16h*)(b0 + k0), (short)0, acc0, false, false);
      acc1 = __builtin_amdgcn_wmma_f32_16x16x32_f16(false, a, false,
                 *(const v16h*)(b1 + k0), (short)0, acc1, false, false);
      acc2 = __builtin_amdgcn_wmma_f32_16x16x32_f16(false, a, false,
                 *(const v16h*)(b2 + k0), (short)0, acc2, false, false);
      acc3 = __builtin_amdgcn_wmma_f32_16x16x32_f16(false, a, false,
                 *(const v16h*)(b3 + k0), (short)0, acc3, false, false);
    }
    #pragma unroll
    for (int vr = 0; vr < 8; ++vr) {
      size_t m = (size_t)(m0 + vr + 8 * hi);
      float* orow = out + m * DOUT + o0 + r;
      orow[ 0] = fmaxf(acc0[vr], 0.f);
      orow[16] = fmaxf(acc1[vr], 0.f);
      orow[32] = fmaxf(acc2[vr], 0.f);
      orow[48] = fmaxf(acc3[vr], 0.f);
    }
  }
}

extern "C" void kernel_launch(void* const* d_in, const int* in_sizes, int n_in,
                              void* d_out, int out_size, void* d_ws, size_t ws_size,
                              hipStream_t stream) {
  const float* feats   = (const float*)d_in[0];   // (8,2048,512) f32
  const float* cw      = (const float*)d_in[1];   // (3,3,512)    f32
  const float* cb      = (const float*)d_in[2];   // (3,3)        f32
  const float* fcw_f32 = (const float*)d_in[3];   // (512,512)    f32
  (void)in_sizes; (void)n_in; (void)out_size; (void)ws_size;
  float* out = (float*)d_out;                     // (8,2048,512) f32

  char* ws = (char*)d_ws;
  size_t off = 0;
  _Float16* xnh    = (_Float16*)(ws + off); off += (size_t)BATCH * NPTS * DIM * 2; // 16 MB
  _Float16* pooled = (_Float16*)(ws + off); off += (size_t)BATCH * NPTS * DIM * 2; // 16 MB
  _Float16* fcwh   = (_Float16*)(ws + off); off += (size_t)DOUT * DIM * 2;         // 0.5 MB
  int*      knnidx = (int*)     (ws + off); off += (size_t)BATCH * NPTS * KNN * 4; // 0.2 MB

  // 1) normalize rows -> f16 Xn                (16384 waves)
  hipLaunchKernelGGL(k_normalize, dim3(BATCH * NPTS / 8), dim3(256), 0, stream,
                     feats, xnh);
  // 2) fc weights -> f16
  hipLaunchKernelGGL(k_fcw_convert, dim3(256), dim3(256), 0, stream,
                     fcw_f32, fcwh);
  // 3) fused Gram GEMM + top-3                 (2048 waves, ILP-4 WMMA)
  hipLaunchKernelGGL(k_sim_topk, dim3(BATCH * (NPTS / 16) / 2), dim3(128), 0, stream,
                     xnh, knnidx);
  // 4) gather + conv + softmax + pool -> f16   (16384 waves)
  hipLaunchKernelGGL(k_conv_pool, dim3(BATCH * NPTS / 8), dim3(256), 0, stream,
                     feats, cw, cb, knnidx, pooled);
  // 5) FC + ReLU                               (2048 waves, ILP-4 WMMA)
  hipLaunchKernelGGL(k_fc_relu, dim3((BATCH * NPTS / 16) / 2), dim3(128), 0, stream,
                     pooled, fcwh, out);
}
